// SimClr_69406671503547
// MI455X (gfx1250) — matrix operand order repaired and checked by reading, not compiled, not run on hardware
//
#include <hip/hip_runtime.h>

typedef __attribute__((ext_vector_type(16))) _Float16 v16h;
typedef __attribute__((ext_vector_type(8)))  _Float16 v8h;
typedef __attribute__((ext_vector_type(8)))  float    v8f;

#define B_ROWS 8192
#define D_DIM  128
#define N_ROWS 16384          // 2*B
#define INV_T  2.0f           // 1 / 0.5
#define EXP_DIAG 7.389056098930650f   // exp(1/T) = e^2, the diagonal of sim
#define LDS_PITCH 136         // halves per LDS row: 128 + 8 pad (272B, bank-conflict-free)
#define BM 256                // rows per workgroup (8 waves x 2 subtiles x 16)

// ---------------------------------------------------------------------------
// Kernel 1: L2-normalize rows of out1/out2 (fp32), emit fp16 Z = [z1; z2]
// and fp32 posdot[i] = <z1_i, z2_i>. One wave (32 lanes) per row pair.
// ---------------------------------------------------------------------------
__global__ __launch_bounds__(256) void simclr_normalize(
    const float* __restrict__ out1, const float* __restrict__ out2,
    _Float16* __restrict__ zh, float* __restrict__ posdot) {
  const int wave = threadIdx.x >> 5;
  const int lane = threadIdx.x & 31;
  const int i = blockIdx.x * 8 + wave;

  const float4 a = ((const float4*)(out1 + (size_t)i * D_DIM))[lane];
  const float4 b = ((const float4*)(out2 + (size_t)i * D_DIM))[lane];

  float s1 = a.x * a.x + a.y * a.y + a.z * a.z + a.w * a.w;
  float s2 = b.x * b.x + b.y * b.y + b.z * b.z + b.w * b.w;
  float sp = a.x * b.x + a.y * b.y + a.z * b.z + a.w * b.w;
#pragma unroll
  for (int m = 16; m >= 1; m >>= 1) {   // wave32 butterfly reduction
    s1 += __shfl_xor(s1, m);
    s2 += __shfl_xor(s2, m);
    sp += __shfl_xor(sp, m);
  }
  const float sc1 = 1.0f / fmaxf(sqrtf(s1), 1e-12f);
  const float sc2 = 1.0f / fmaxf(sqrtf(s2), 1e-12f);

  _Float16* z1p = zh + (size_t)i * D_DIM + 4 * lane;
  _Float16* z2p = zh + (size_t)(B_ROWS + i) * D_DIM + 4 * lane;
  z1p[0] = (_Float16)(a.x * sc1); z1p[1] = (_Float16)(a.y * sc1);
  z1p[2] = (_Float16)(a.z * sc1); z1p[3] = (_Float16)(a.w * sc1);
  z2p[0] = (_Float16)(b.x * sc2); z2p[1] = (_Float16)(b.y * sc2);
  z2p[2] = (_Float16)(b.z * sc2); z2p[3] = (_Float16)(b.w * sc2);

  if (lane == 0) posdot[i] = sp * sc1 * sc2;
}

// Build a v16h from two 8-half chunks
__device__ __forceinline__ v16h cat16(v8h lo, v8h hi) {
  return __builtin_shufflevector(lo, hi, 0, 1, 2, 3, 4, 5, 6, 7,
                                         8, 9, 10, 11, 12, 13, 14, 15);
}

// ---------------------------------------------------------------------------
// Kernel 2: for each row i of Z (16384x128 fp16), compute
//   denom_i = sum_j exp(<z_i, z_j>/T) - e^2
//   lossrow[i] = log(denom_i) - posdot[i mod B]/T
// WG = 256 rows, 8 waves; wave w owns 2 subtiles of 16 rows each, so every
// B-register fetch from LDS feeds TWO wmma ops (ds:wmma = 1:1) and the two
// accumulator chains hide WMMA RAW latency.
// ---------------------------------------------------------------------------
__global__ __launch_bounds__(256) void simclr_rowsum(
    const _Float16* __restrict__ zh, const float* __restrict__ posdot,
    float* __restrict__ lossrow) {
  __shared__ _Float16 bt[128 * LDS_PITCH];   // 34 KB, padded rows

  const int tid  = threadIdx.x;
  const int wave = tid >> 5;
  const int lane = tid & 31;
  const int lh   = lane >> 4;     // lane-half (0: lanes 0-15, 1: lanes 16-31)
  const int idx  = lane & 15;
  const int i0   = blockIdx.x * BM;

  // --- Preload this wave's two A subtiles (2 x 16 rows x 128 K).
  // ISA 16-bit A 16x32 layout: lane idx = row M; element e of v16h holds
  // K = 32c + (e<8 ? 8*lh + e : 16 + 8*lh + (e-8)).
  v16h A[2][4];
#pragma unroll
  for (int s = 0; s < 2; ++s) {
    const int arow = i0 + wave * 32 + s * 16 + idx;
    const _Float16* ap = zh + (size_t)arow * D_DIM;
#pragma unroll
    for (int c = 0; c < 4; ++c) {
      v8h lo = *(const v8h*)(ap + 32 * c + 8 * lh);
      v8h hi = *(const v8h*)(ap + 32 * c + 16 + 8 * lh);
      A[s][c] = cat16(lo, hi);
    }
  }

  v8f rs0 = {}, rs1 = {};   // persistent per-lane partial row-sums of exp

#pragma unroll 1
  for (int j0 = 0; j0 < N_ROWS; j0 += 128) {
    __syncthreads();   // previous tile fully consumed
    // Stage 128x128 fp16 B tile: 2048 x 16B, 8 per thread.
#pragma unroll
    for (int q = tid; q < 2048; q += 256) {
      const int r = q >> 4, c16 = q & 15;
      const uint4 v = ((const uint4*)(zh + (size_t)(j0 + r) * D_DIM))[c16];
      ((uint4*)(&bt[r * LDS_PITCH]))[c16] = v;
    }
    __syncthreads();

#pragma unroll
    for (int t = 0; t < 8; ++t) {
      // B-matrix 32x16 layout: lane idx = column N; element e holds
      // K = 32k + (e<8 ? 8*lh + e : 16 + 8*lh + (e-8)), matching A.
      // Load ALL 4 K-chunks into distinct regs first so the scheduler can
      // batch the ds_loads, wait once, then run the wmma burst.
      const _Float16* bp = &bt[(t * 16 + idx) * LDS_PITCH];
      v16h Bv[4];
#pragma unroll
      for (int k = 0; k < 4; ++k) {
        v8h lo = *(const v8h*)(bp + 32 * k + 8 * lh);
        v8h hi = *(const v8h*)(bp + 32 * k + 16 + 8 * lh);
        Bv[k] = cat16(lo, hi);
      }
      v8f c0 = {}, c1 = {};
#pragma unroll
      for (int k = 0; k < 4; ++k) {
        c0 = __builtin_amdgcn_wmma_f32_16x16x32_f16(
            false, A[0][k], false, Bv[k], (short)0, c0, false, false);
        c1 = __builtin_amdgcn_wmma_f32_16x16x32_f16(
            false, A[1][k], false, Bv[k], (short)0, c1, false, false);
      }
#pragma unroll
      for (int e = 0; e < 8; ++e) {
        rs0[e] += __expf(INV_T * c0[e]);
        rs1[e] += __expf(INV_T * c1[e]);
      }
    }
  }

  // C layout: VGPR e holds (M = 8*lh + e, N = idx). Reduce over the 16
  // lanes of each column group so lanes idx==0 hold full row sums.
#pragma unroll
  for (int m = 1; m < 16; m <<= 1) {
#pragma unroll
    for (int e = 0; e < 8; ++e) {
      rs0[e] += __shfl_xor(rs0[e], m);
      rs1[e] += __shfl_xor(rs1[e], m);
    }
  }

  if (idx == 0) {
#pragma unroll
    for (int e = 0; e < 8; ++e) {
      const int row0 = i0 + wave * 32 + 8 * lh + e;        // subtile 0
      const int row1 = row0 + 16;                           // subtile 1
      lossrow[row0] = __logf(rs0[e] - EXP_DIAG) - INV_T * posdot[row0 & (B_ROWS - 1)];
      lossrow[row1] = __logf(rs1[e] - EXP_DIAG) - INV_T * posdot[row1 & (B_ROWS - 1)];
    }
  }
}

// ---------------------------------------------------------------------------
// Kernel 3: deterministic fixed-order reduction of the 16384 row losses.
// ---------------------------------------------------------------------------
__global__ __launch_bounds__(256) void simclr_finalize(
    const float* __restrict__ lossrow, float* __restrict__ out) {
  __shared__ float red[256];
  float s = 0.0f;
  for (int i = threadIdx.x; i < N_ROWS; i += 256) s += lossrow[i];
  red[threadIdx.x] = s;
  __syncthreads();
  for (int off = 128; off > 0; off >>= 1) {
    if (threadIdx.x < off) red[threadIdx.x] += red[threadIdx.x + off];
    __syncthreads();
  }
  if (threadIdx.x == 0) out[0] = red[0] / (float)N_ROWS;
}

extern "C" void kernel_launch(void* const* d_in, const int* in_sizes, int n_in,
                              void* d_out, int out_size, void* d_ws, size_t ws_size,
                              hipStream_t stream) {
  (void)in_sizes; (void)n_in; (void)out_size; (void)ws_size;
  const float* out1 = (const float*)d_in[0];
  const float* out2 = (const float*)d_in[1];
  // d_in[2] (indices) is unused by the loss.

  // Workspace: lossrow[16384] f32 | posdot[8192] f32 | zh[16384*128] f16 (~4.2MB)
  float*    lossrow = (float*)d_ws;
  float*    posdot  = lossrow + N_ROWS;
  _Float16* zh      = (_Float16*)(posdot + B_ROWS);

  simclr_normalize<<<B_ROWS / 8, 256, 0, stream>>>(out1, out2, zh, posdot);
  simclr_rowsum<<<N_ROWS / BM, 256, 0, stream>>>(zh, posdot, lossrow);
  simclr_finalize<<<1, 256, 0, stream>>>(lossrow, (float*)d_out);
}
